// GroupedQueryAttention_63883343561228
// MI455X (gfx1250) — matrix operand order repaired and checked by reading, not compiled, not run on hardware
//
#include <hip/hip_runtime.h>

typedef __attribute__((ext_vector_type(16))) __bf16 v16bf;
typedef __attribute__((ext_vector_type(8)))  __bf16 v8bf;
typedef __attribute__((ext_vector_type(8)))  float  v8f;

#define D_MODEL 1024
#define L_SEQ   4096
#define NHEAD   16
#define KVDIM   256   // 4 kv heads * 64
#define HEAD_DIM 64

static __device__ __forceinline__ __bf16 f2bf(float f) {
  union { float f; unsigned u; } in; in.f = f;
  unsigned u = in.u;
  unsigned r = u + 0x7fffu + ((u >> 16) & 1u);   // round-to-nearest-even
  union { unsigned short s; __bf16 b; } out;
  out.s = (unsigned short)(r >> 16);
  return out.b;
}

// xor-shuffle within each 16-lane row. Prefer v_permlane16_b32 (VALU) over
// ds_bpermute (DS op + dscnt wait). sel patterns: nibble i selects lane (i^mask).
static __device__ __forceinline__ float rowshuffle(float v, int mask, unsigned lo, unsigned hi) {
#if __has_builtin(__builtin_amdgcn_permlane16)
  unsigned u = __builtin_bit_cast(unsigned, v);
  u = __builtin_amdgcn_permlane16(u, u, lo, hi, false, false);
  return __builtin_bit_cast(float, u);
#elif __has_builtin(__builtin_amdgcn_permlane16_b32)
  unsigned u = __builtin_bit_cast(unsigned, v);
  u = __builtin_amdgcn_permlane16_b32(u, u, lo, hi, false, false);
  return __builtin_bit_cast(float, u);
#else
  return __shfl_xor(v, mask, 32);
#endif
}

static __device__ __forceinline__ float rowmax16(float v) {
  v = fmaxf(v, rowshuffle(v, 1, 0x67452301u, 0xEFCDAB89u));
  v = fmaxf(v, rowshuffle(v, 2, 0x54761032u, 0xDCFE98BAu));
  v = fmaxf(v, rowshuffle(v, 4, 0x32107654u, 0xBA98FEDCu));
  v = fmaxf(v, rowshuffle(v, 8, 0xFEDCBA98u, 0x76543210u));
  return v;
}

// ---------------- conversion kernels ----------------
__global__ void cvt_f32_bf16(const float* __restrict__ in, __bf16* __restrict__ out, int n) {
  int i = blockIdx.x * blockDim.x + threadIdx.x;
  if (i < n) out[i] = f2bf(in[i]);
}

// W [K x N] fp32 row-major -> Wt [N x K] bf16 (transposed: B-fragments contiguous)
__global__ void cvt_w_transpose(const float* __restrict__ W, __bf16* __restrict__ Wt, int K, int N) {
  int i = blockIdx.x * blockDim.x + threadIdx.x;
  if (i >= K * N) return;
  int n = i / K, k = i % K;
  Wt[i] = f2bf(W[(size_t)k * N + n]);
}

// V [pos][c] bf16 -> Vt [c][pos] bf16   (c = kv*64 + d)
__global__ void transpose_v(const __bf16* __restrict__ V, __bf16* __restrict__ Vt) {
  int idx = blockIdx.x * blockDim.x + threadIdx.x;
  if (idx >= L_SEQ * KVDIM) return;
  int pos = idx >> 8;
  int c   = idx & 255;
  Vt[(size_t)c * L_SEQ + pos] = V[idx];
}

// ---------------- WMMA GEMM: out[M x N] = oscale*(A[M x K] @ Bt[N x K]^T + bias) ----
// block = 128 threads (4 waves); block tile 64 x 64; wave tile 16 x 64.
// Register double-buffered: fragments for k+32 load while WMMAs for k issue.
template <bool F32OUT>
__global__ void gemm_wmma(const __bf16* __restrict__ A, const __bf16* __restrict__ Bt,
                          const float* __restrict__ bias, void* __restrict__ out,
                          int K, int N, float oscale) {
  const int lane = threadIdx.x & 31;
  const int w    = threadIdx.x >> 5;
  const int hsel = (lane >> 4) & 1;
  const int ncol = lane & 15;
  const int kb8  = hsel * 8;
  const int kb16 = hsel * 16;
  const int nbase = blockIdx.y * 64;
  const int arow  = blockIdx.x * 64 + w * 16 + ncol;

  union AB { v16bf v; v8bf h8[2]; };
  const __bf16* ap = A + (size_t)arow * K;

  auto loadA = [&](int k0, AB& a) {
    a.v = *(const v16bf*)(ap + k0 + kb16);
  };
  auto loadB = [&](int k0, v16bf* b) {
#pragma unroll
    for (int nt = 0; nt < 4; ++nt)
      b[nt] = *(const v16bf*)(Bt + (size_t)(nbase + nt * 16 + ncol) * K + k0 + kb16);
  };

  v8f acc[4] = {};
  AB a_cur, a_nxt;
  v16bf b_cur[4], b_nxt[4];
  loadA(0, a_cur);
  loadB(0, b_cur);

  for (int k0 = 0; k0 < K; k0 += 32) {
    const bool more = (k0 + 32) < K;
    if (more) { loadA(k0 + 32, a_nxt); loadB(k0 + 32, b_nxt); }
#pragma unroll
    for (int nt = 0; nt < 4; ++nt)
      acc[nt] = __builtin_amdgcn_wmma_f32_16x16x32_bf16(false, a_cur.v, false, b_cur[nt],
                                                        (short)0, acc[nt], false, false);
    if (more) {
      a_cur = a_nxt;
#pragma unroll
      for (int nt = 0; nt < 4; ++nt) b_cur[nt] = b_nxt[nt];
    }
  }

  const int rowbase = blockIdx.x * 64 + w * 16 + hsel * 8;
#pragma unroll
  for (int nt = 0; nt < 4; ++nt) {
    int n = nbase + nt * 16 + ncol;
    float bv = bias[n];
#pragma unroll
    for (int r = 0; r < 8; ++r) {
      float v = (acc[nt][r] + bv) * oscale;
      size_t idx = (size_t)(rowbase + r) * N + n;
      if constexpr (F32OUT) ((float*)out)[idx] = v;
      else                  ((__bf16*)out)[idx] = f2bf(v);
    }
  }
}

// ---------------- flash attention ----------------
// grid = (L/64, NHEAD); block = 128 (4 waves). All 4 waves share one head, so
// K/V tiles are staged cooperatively into LDS with GLOBAL_LOAD_ASYNC_TO_LDS_B128
// (ASYNCcnt DMA), double-buffered: wait(tile j) -> barrier -> async(tile j+1)
// -> compute(tile j). Q is pre-scaled by 1/sqrt(HEAD_DIM) at projection time.
__global__ void flash_attn(const __bf16* __restrict__ Q, const __bf16* __restrict__ Kb,
                           const __bf16* __restrict__ Vt, __bf16* __restrict__ O) {
  __shared__ __align__(64) __bf16 kt[2][32][64];     // [buf][key][d]       8 KB
  __shared__ __align__(64) __bf16 vtile[2][64][32];  // [buf][d][pos]       8 KB
  __shared__ __align__(64) __bf16 pbuf[4][16][32];   // per-wave P tile     4 KB

  const int lane = threadIdx.x & 31;
  const int w    = threadIdx.x >> 5;
  const int hsel = (lane >> 4) & 1;
  const int ncol = lane & 15;
  const int kb8  = hsel * 8;
  const int kb16 = hsel * 16;
  const int h    = blockIdx.y;
  const int kv   = h >> 2;
  const int qrow0 = blockIdx.x * 64 + w * 16;

  // cooperative async stage of one 32-key K/V tile into LDS buffer b
  auto stage = [&](int b, int j0s) {
    {
      int key  = threadIdx.x >> 2;            // 0..31
      int doff = (threadIdx.x & 3) * 16;      // 0,16,32,48
      const __bf16* g = Kb + (size_t)(j0s + key) * KVDIM + kv * HEAD_DIM + doff;
      unsigned l = (unsigned)(uintptr_t)&kt[b][key][doff];
      asm volatile("global_load_async_to_lds_b128 %0, %1, off\n\t"
                   "global_load_async_to_lds_b128 %0, %1, off offset:16"
                   :: "v"(l), "v"(g) : "memory");
    }
    {
      int d    = threadIdx.x >> 1;            // 0..63
      int poff = (threadIdx.x & 1) * 16;      // 0,16
      const __bf16* g = Vt + (size_t)(kv * HEAD_DIM + d) * L_SEQ + j0s + poff;
      unsigned l = (unsigned)(uintptr_t)&vtile[b][d][poff];
      asm volatile("global_load_async_to_lds_b128 %0, %1, off\n\t"
                   "global_load_async_to_lds_b128 %0, %1, off offset:16"
                   :: "v"(l), "v"(g) : "memory");
    }
  };

  union F16 { v16bf v; v8bf h8[2]; };

  // Q fragments (loaded once from global; already scaled by 1/8)
  F16 aq[2];
  {
    const __bf16* qp = Q + (size_t)(qrow0 + ncol) * D_MODEL + h * HEAD_DIM;
#pragma unroll
    for (int c = 0; c < 2; ++c) {
      aq[c].v = *(const v16bf*)(qp + 32 * c + kb16);
    }
  }

  v16bf bones;   // 32x16 all-ones B matrix -> row sums of P via WMMA
#pragma unroll
  for (int i = 0; i < 16; ++i) bones[i] = f2bf(1.0f);

  v8f o[4] = {};
  v8f ol   = {};                 // running row-sum accumulator (flash l)
  float mrow[8];
#pragma unroll
  for (int r = 0; r < 8; ++r) mrow[r] = -3.0e38f;

  const int NB = L_SEQ / 32;

  auto body = [&](int jb, int buf) {
    asm volatile("s_wait_asynccnt 0x0" ::: "memory");   // tile jb landed in LDS
    __syncthreads();                                    // visible to all waves
    if (jb + 1 < NB) stage(buf ^ 1, (jb + 1) * 32);     // DMA next tile under compute

    // ---- S = Q @ K^T : two 16x16 tiles, K-dim 64 = 2 wmma steps each ----
    v8f s[2] = {};
#pragma unroll
    for (int t = 0; t < 2; ++t) {
#pragma unroll
      for (int c = 0; c < 2; ++c) {
        v16bf b = *(const v16bf*)&kt[buf][16 * t + ncol][32 * c + kb16];
        s[t] = __builtin_amdgcn_wmma_f32_16x16x32_bf16(false, aq[c].v, false, b,
                                                       (short)0, s[t], false, false);
      }
    }

    // ---- online softmax: row max via permlane16 VALU shuffles ----
#pragma unroll
    for (int r = 0; r < 8; ++r) {
      float s0 = s[0][r];
      float s1 = s[1][r];
      float mx = rowmax16(fmaxf(s0, s1));
      float mnew = fmaxf(mrow[r], mx);
      float corr = __expf(mrow[r] - mnew);
      mrow[r] = mnew;
      float p0 = __expf(s0 - mnew);
      float p1 = __expf(s1 - mnew);
#pragma unroll
      for (int nt = 0; nt < 4; ++nt) o[nt][r] *= corr;
      ol[r] *= corr;
      int mr = r + hsel * 8;
      pbuf[w][mr][ncol]      = f2bf(p0);
      pbuf[w][mr][ncol + 16] = f2bf(p1);
    }

    // ---- O += P @ V ; l += P @ ones ----
    F16 aprow;
    aprow.v = *(const v16bf*)&pbuf[w][ncol][kb16];
#pragma unroll
    for (int nt = 0; nt < 4; ++nt) {
      v16bf b = *(const v16bf*)&vtile[buf][nt * 16 + ncol][kb16];
      o[nt] = __builtin_amdgcn_wmma_f32_16x16x32_bf16(false, aprow.v, false, b,
                                                      (short)0, o[nt], false, false);
    }
    ol = __builtin_amdgcn_wmma_f32_16x16x32_bf16(false, aprow.v, false, bones,
                                                 (short)0, ol, false, false);
  };

  stage(0, 0);
  for (int jb = 0; jb < NB; jb += 2) {   // 2x unrolled: literal LDS buffer indices
    body(jb, 0);
    body(jb + 1, 1);
  }

  // ---- epilogue: scale rows by 1/l, store bf16 ----
#pragma unroll
  for (int r = 0; r < 8; ++r) {
    float invl = 1.0f / ol[r];
    int m = qrow0 + hsel * 8 + r;
#pragma unroll
    for (int nt = 0; nt < 4; ++nt) {
      O[(size_t)m * D_MODEL + h * HEAD_DIM + nt * 16 + ncol] = f2bf(o[nt][r] * invl);
    }
  }
}

// ---------------- launch ----------------
extern "C" void kernel_launch(void* const* d_in, const int* in_sizes, int n_in,
                              void* d_out, int out_size, void* d_ws, size_t ws_size,
                              hipStream_t stream) {
  const float* x  = (const float*)d_in[0];
  const float* Wq = (const float*)d_in[1];
  const float* bq = (const float*)d_in[2];
  const float* Wk = (const float*)d_in[3];
  const float* bk = (const float*)d_in[4];
  const float* Wv = (const float*)d_in[5];
  const float* bv = (const float*)d_in[6];
  const float* Wo = (const float*)d_in[7];
  const float* bo = (const float*)d_in[8];
  float* out = (float*)d_out;

  char* ws = (char*)d_ws;
  size_t off = 0;
  auto alloc = [&](size_t bytes) -> char* {
    char* p = ws + off;
    off += (bytes + 255) & ~(size_t)255;
    return p;
  };
  __bf16* xb  = (__bf16*)alloc((size_t)L_SEQ * D_MODEL * 2);
  __bf16* Wqt = (__bf16*)alloc((size_t)D_MODEL * D_MODEL * 2);
  __bf16* Wkt = (__bf16*)alloc((size_t)KVDIM * D_MODEL * 2);
  __bf16* Wvt = (__bf16*)alloc((size_t)KVDIM * D_MODEL * 2);
  __bf16* Wot = (__bf16*)alloc((size_t)D_MODEL * D_MODEL * 2);
  __bf16* Qb  = (__bf16*)alloc((size_t)L_SEQ * D_MODEL * 2);
  __bf16* Kbf = (__bf16*)alloc((size_t)L_SEQ * KVDIM * 2);
  __bf16* Vb  = (__bf16*)alloc((size_t)L_SEQ * KVDIM * 2);
  __bf16* Vt  = (__bf16*)alloc((size_t)KVDIM * L_SEQ * 2);
  __bf16* Ob  = (__bf16*)alloc((size_t)L_SEQ * D_MODEL * 2);

  cvt_f32_bf16<<<(L_SEQ * D_MODEL + 255) / 256, 256, 0, stream>>>(x, xb, L_SEQ * D_MODEL);
  cvt_w_transpose<<<(D_MODEL * D_MODEL + 255) / 256, 256, 0, stream>>>(Wq, Wqt, D_MODEL, D_MODEL);
  cvt_w_transpose<<<(D_MODEL * KVDIM + 255) / 256, 256, 0, stream>>>(Wk, Wkt, D_MODEL, KVDIM);
  cvt_w_transpose<<<(D_MODEL * KVDIM + 255) / 256, 256, 0, stream>>>(Wv, Wvt, D_MODEL, KVDIM);
  cvt_w_transpose<<<(D_MODEL * D_MODEL + 255) / 256, 256, 0, stream>>>(Wo, Wot, D_MODEL, D_MODEL);

  // Q is pre-scaled by 1/sqrt(HEAD_DIM) so attention skips the per-score multiply
  gemm_wmma<false><<<dim3(L_SEQ / 64, D_MODEL / 64), 128, 0, stream>>>(xb, Wqt, bq, Qb, D_MODEL, D_MODEL, 0.125f);
  gemm_wmma<false><<<dim3(L_SEQ / 64, KVDIM / 64), 128, 0, stream>>>(xb, Wkt, bk, Kbf, D_MODEL, KVDIM, 1.0f);
  gemm_wmma<false><<<dim3(L_SEQ / 64, KVDIM / 64), 128, 0, stream>>>(xb, Wvt, bv, Vb, D_MODEL, KVDIM, 1.0f);

  transpose_v<<<(L_SEQ * KVDIM + 255) / 256, 256, 0, stream>>>(Vb, Vt);

  flash_attn<<<dim3(L_SEQ / 64, NHEAD), 128, 0, stream>>>(Qb, Kbf, Vt, Ob);

  gemm_wmma<true><<<dim3(L_SEQ / 64, D_MODEL / 64), 128, 0, stream>>>(Ob, Wot, bo, out, D_MODEL, D_MODEL, 1.0f);
}
